// GRUCell_58402965291333
// MI455X (gfx1250) — matrix operand order repaired and checked by reading, not compiled, not run on hardware
//
#include <hip/hip_runtime.h>
#include <hip/hip_bf16.h>

// ---------------------------------------------------------------------------
// GRU cell on MI455X (gfx1250):
//   concat = [h, x]                        (16384 x 2048)
//   r = sigmoid(concat @ W_r)              (GEMM 1, epilogue writes bf16(r*h))
//   z = sigmoid(concat @ W_z)              (GEMM 2, parked in d_out as f32)
//   h~ = tanh([r*h, x] @ W_h)              (GEMM 3, fused final blend)
//   out = (1-z)*h + z*h~
// GEMMs use v_wmma_f32_16x16x32_bf16 (wave32), f32 accumulate.
// Tile staging uses GLOBAL_LOAD_ASYNC_TO_LDS_B128 (ASYNCcnt) when available.
// ---------------------------------------------------------------------------

#define BATCH   16384
#define HID     1024
#define KDIM    2048         // hidden + input
#define KSTEP   32           // K per WMMA
#define NKITER  (KDIM / KSTEP)
#define LDT     40           // LDS tile K-stride in halfs (32 + 8 pad)

typedef __attribute__((ext_vector_type(16))) __bf16 v16bf;
typedef __attribute__((ext_vector_type(8)))  __bf16 v8bf;
typedef __attribute__((ext_vector_type(4)))  __bf16 v4bf;
typedef __attribute__((ext_vector_type(8)))  float  v8f;
typedef __attribute__((ext_vector_type(4)))  int    v4i;

// addrspace-qualified pointee typedefs for the async-copy builtin
typedef __attribute__((address_space(1))) v4i* as1_v4i_ptr;  // global
typedef __attribute__((address_space(3))) v4i* as3_v4i_ptr;  // LDS

#if __has_builtin(__builtin_amdgcn_global_load_async_to_lds_b128) && \
    __has_builtin(__builtin_amdgcn_s_wait_asynccnt)
#define USE_ASYNC 1
#else
#define USE_ASYNC 0
#endif

// 16-byte global -> LDS copy (async on gfx1250 when builtin available)
__device__ __forceinline__ void cp16_g2l(const __bf16* g, __bf16* l)
{
#if USE_ASYNC
    __builtin_amdgcn_global_load_async_to_lds_b128(
        (as1_v4i_ptr)const_cast<__bf16*>(g),
        (as3_v4i_ptr)l,
        0, 0);
#else
    *(uint4*)l = *(const uint4*)g;
#endif
}

__device__ __forceinline__ void wait_async_copies()
{
#if USE_ASYNC
    __builtin_amdgcn_s_wait_asynccnt(0);
#endif
}

// ---------------------------------------------------------------------------
// f32 -> bf16 conversion, 4 elements/thread
// ---------------------------------------------------------------------------
__global__ __launch_bounds__(256) void cvt_f32_to_bf16(
    const float* __restrict__ src, __bf16* __restrict__ dst, int n4)
{
    int i = blockIdx.x * blockDim.x + threadIdx.x;
    if (i < n4) {
        float4 v = ((const float4*)src)[i];
        v4bf o = { (__bf16)v.x, (__bf16)v.y, (__bf16)v.z, (__bf16)v.w };
        *(v4bf*)(dst + (size_t)i * 4) = o;
    }
}

// ---------------------------------------------------------------------------
// W [K=2048][N=1024] f32 row-major  ->  Wt [N][K] bf16  (LDS-tiled transpose)
// ---------------------------------------------------------------------------
__global__ __launch_bounds__(256) void transpose_w_bf16(
    const float* __restrict__ W, __bf16* __restrict__ Wt, int K, int N)
{
    __shared__ float tile[32][33];
    const int bx = blockIdx.x * 32;           // N base
    const int by = blockIdx.y * 32;           // K base
    const int tx = threadIdx.x & 31;
    const int ty = threadIdx.x >> 5;          // 0..7
#pragma unroll
    for (int i = ty; i < 32; i += 8)
        tile[i][tx] = W[(size_t)(by + i) * N + bx + tx];   // coalesced over N
    __syncthreads();
#pragma unroll
    for (int i = ty; i < 32; i += 8)
        Wt[(size_t)(bx + i) * K + by + tx] = (__bf16)tile[tx][i]; // coalesced over K
}

// ---------------------------------------------------------------------------
// Tiled bf16 GEMM: C[16384 x 1024] = [A0 | A1] (16384 x 2048) @ Wt^T
//   A0, A1 : bf16 [BATCH][1024]  (K 0..1023 from A0, 1024..2047 from A1)
//   Wt     : bf16 [N=1024][K=2048]   (pre-transposed weights)
// Block: 256 thr = 8 waves; tile 128(M) x 128(N); each wave: 4x2 16x16 tiles.
// MODE 0: out_bf  = bf16( sigmoid(acc) * h )        (r-gate, produces r*h)
// MODE 1: out_f   = sigmoid(acc)                    (z-gate, parked in d_out)
// MODE 2: out_f   = (1-z)*h + z*tanh(acc)           (final blend)
// ---------------------------------------------------------------------------
template <int MODE>
__global__ __launch_bounds__(256) void gru_gemm(
    const __bf16* __restrict__ A0, const __bf16* __restrict__ A1,
    const __bf16* __restrict__ Wt,
    const float* __restrict__ hprev,
    const float* zin,            // MODE 2 only (aliases out_f)
    __bf16* __restrict__ out_bf, // MODE 0 only
    float* out_f)                // MODE 1/2 only
{
    __shared__ __bf16 lA[2][128 * LDT];
    __shared__ __bf16 lB[2][128 * LDT];

    const int tid  = threadIdx.x;
    const int lane = tid & 31;
    const int wave = tid >> 5;              // 0..7
    const int wm   = (wave & 1) * 64;       // wave M origin inside block tile
    const int wn   = (wave >> 1) * 32;      // wave N origin inside block tile
    const int lm   = lane & 15;
    const int hi16 = lane >> 4;             // 0 or 1
    const int ga   = hi16 * 8;              // A-frag K sub-offset (halfs)
    const int kb   = hi16 * 16;             // unused; B-frag K sub-offset is ga

    const int srow = tid >> 1;              // staging row 0..127
    const int scol = (tid & 1) * 16;        // staging col (halfs): 0 or 16

    const int m0 = blockIdx.x * 128;
    const int n0 = blockIdx.y * 128;

    v8f acc[4][2] = {};

    // Issue one K-tile's worth of global->LDS copies (4 x B128 per thread)
    auto stage = [&](int kt, int c) {
        const int kk = kt * KSTEP;
        const __bf16* Asrc = (kk < HID) ? A0 : A1;
        const int ac = (kk < HID) ? kk : (kk - HID);
        const __bf16* ap = Asrc + (size_t)(m0 + srow) * HID + ac + scol;
        __bf16*       al = &lA[c][srow * LDT + scol];
        const __bf16* bp = Wt + (size_t)(n0 + srow) * KDIM + kk + scol;
        __bf16*       bl = &lB[c][srow * LDT + scol];
        cp16_g2l(ap,     al);
        cp16_g2l(ap + 8, al + 8);
        cp16_g2l(bp,     bl);
        cp16_g2l(bp + 8, bl + 8);
    };

    stage(0, 0);
    wait_async_copies();
    __syncthreads();

    int cur = 0;
    for (int kt = 0; kt < NKITER; ++kt) {
        if (kt + 1 < NKITER) stage(kt + 1, cur ^ 1);   // async, overlaps WMMAs

        v16bf af[4], bfr[2];
#pragma unroll
        for (int mi = 0; mi < 4; ++mi) {
            // ISA 16-bit A layout: lane holds K[ga..ga+7] and K[16+ga..16+ga+7]
            const __bf16* ar = &lA[cur][(wm + mi * 16 + lm) * LDT];
            v8bf lo = *(const v8bf*)(ar + ga);
            v8bf hi = *(const v8bf*)(ar + 16 + ga);
            af[mi] = __builtin_shufflevector(lo, hi,
                0, 1, 2, 3, 4, 5, 6, 7, 8, 9, 10, 11, 12, 13, 14, 15);
        }
#pragma unroll
        for (int ni = 0; ni < 2; ++ni) {
            // ISA 16-bit B layout: lane = N%16, K[ga..ga+7] and K[16+ga..16+ga+7]
            const __bf16* br = &lB[cur][(wn + ni * 16 + lm) * LDT + ga];
            v8bf lo = *(const v8bf*)(br);
            v8bf hi = *(const v8bf*)(br + 16);
            bfr[ni] = __builtin_shufflevector(lo, hi,
                0, 1, 2, 3, 4, 5, 6, 7, 8, 9, 10, 11, 12, 13, 14, 15);
        }
#pragma unroll
        for (int mi = 0; mi < 4; ++mi)
#pragma unroll
            for (int ni = 0; ni < 2; ++ni)
                acc[mi][ni] = __builtin_amdgcn_wmma_f32_16x16x32_bf16(
                    false, af[mi], false, bfr[ni],
                    (short)0, acc[mi][ni], false, false);

        if (kt + 1 < NKITER) wait_async_copies();      // prefetch landed
        __syncthreads();
        cur ^= 1;
    }

    // Epilogue — C/D layout: lane = N%16, row = vgpr + 8*(lane>=16)
#pragma unroll
    for (int mi = 0; mi < 4; ++mi) {
#pragma unroll
        for (int ni = 0; ni < 2; ++ni) {
#pragma unroll
            for (int r = 0; r < 8; ++r) {
                const int row = m0 + wm + mi * 16 + hi16 * 8 + r;
                const int col = n0 + wn + ni * 16 + lm;
                const size_t idx = (size_t)row * HID + col;
                const float a = acc[mi][ni][r];
                if (MODE == 0) {
                    const float s = 1.0f / (1.0f + __expf(-a));
                    out_bf[idx] = (__bf16)(s * hprev[idx]);
                } else if (MODE == 1) {
                    out_f[idx] = 1.0f / (1.0f + __expf(-a));
                } else {
                    const float zv = zin[idx];
                    const float hv = hprev[idx];
                    const float ht = tanhf(a);
                    out_f[idx] = (1.0f - zv) * hv + zv * ht;
                }
            }
        }
    }
}

// ---------------------------------------------------------------------------
extern "C" void kernel_launch(void* const* d_in, const int* in_sizes, int n_in,
                              void* d_out, int out_size, void* d_ws, size_t ws_size,
                              hipStream_t stream)
{
    const float* h  = (const float*)d_in[0];
    const float* x  = (const float*)d_in[1];
    const float* Wr = (const float*)d_in[2];
    const float* Wz = (const float*)d_in[3];
    const float* Wh = (const float*)d_in[4];
    float* out = (float*)d_out;

    char* ws = (char*)d_ws;
    __bf16* h_bf  = (__bf16*)ws;  ws += (size_t)BATCH * HID * 2;   // 32 MiB
    __bf16* x_bf  = (__bf16*)ws;  ws += (size_t)BATCH * HID * 2;   // 32 MiB
    __bf16* rh_bf = (__bf16*)ws;  ws += (size_t)BATCH * HID * 2;   // 32 MiB
    __bf16* wr_t  = (__bf16*)ws;  ws += (size_t)KDIM * HID * 2;    //  4 MiB
    __bf16* wz_t  = (__bf16*)ws;  ws += (size_t)KDIM * HID * 2;    //  4 MiB
    __bf16* wh_t  = (__bf16*)ws;  ws += (size_t)KDIM * HID * 2;    //  4 MiB

    // 1) precision conversion (bandwidth-side, one pass)
    {
        const int n4 = BATCH * HID / 4;
        cvt_f32_to_bf16<<<n4 / 256, 256, 0, stream>>>(h, h_bf, n4);
        cvt_f32_to_bf16<<<n4 / 256, 256, 0, stream>>>(x, x_bf, n4);
        dim3 tg(HID / 32, KDIM / 32);
        transpose_w_bf16<<<tg, 256, 0, stream>>>(Wr, wr_t, KDIM, HID);
        transpose_w_bf16<<<tg, 256, 0, stream>>>(Wz, wz_t, KDIM, HID);
        transpose_w_bf16<<<tg, 256, 0, stream>>>(Wh, wh_t, KDIM, HID);
    }

    // 2) three WMMA GEMMs
    dim3 grid(BATCH / 128, HID / 128);
    gru_gemm<0><<<grid, 256, 0, stream>>>(h_bf, x_bf, wr_t, h, nullptr, rh_bf, nullptr);
    gru_gemm<1><<<grid, 256, 0, stream>>>(h_bf, x_bf, wz_t, h, nullptr, nullptr, out);
    gru_gemm<2><<<grid, 256, 0, stream>>>(rh_bf, x_bf, wh_t, h, out, nullptr, out);
}
